// MinimalRNNCell_5351529251117
// MI455X (gfx1250) — matrix operand order repaired and checked by reading, not compiled, not run on hardware
//
#include <hip/hip_runtime.h>

typedef __attribute__((ext_vector_type(16))) __bf16        v16bf;
typedef __attribute__((ext_vector_type(8)))  float         v8f;
typedef __attribute__((ext_vector_type(4)))  unsigned int  v4u;
typedef __attribute__((ext_vector_type(4)))  float         v4f;

namespace {
constexpr int Bn = 256;   // batch
constexpr int Tn = 2048;  // time
constexpr int Un = 128;   // units (= D)
constexpr int Ln = 32;    // chunk length
constexpr int Cn = Tn / Ln; // 64 chunks
constexpr int ROWW = 68;  // LDS row stride in 32-bit words (128 bf16 + pad, 16B aligned, bank-conflict free)

union Frag {
  unsigned int u[8];
  v4u          q[2];
  v16bf        b;
};
} // namespace

__device__ __forceinline__ unsigned int pack2f(float lo, float hi) {
  union { __bf16 h; unsigned short s; } a, b;
  a.h = (__bf16)lo; b.h = (__bf16)hi;
  return (unsigned int)a.s | ((unsigned int)b.s << 16);
}

// ---------------------------------------------------------------------------
// Pass A: per-chunk fused scan with zero entering state.
//   h_local[t] = x[t]@K + h_local[t-1]@R  within chunk; writes local states.
// One WG = 16 batch rows x one chunk. 8 waves; wave w owns output cols [16w,16w+16).
// ---------------------------------------------------------------------------
__global__ __launch_bounds__(256, 1)
void rnn_chunk_scan(const float* __restrict__ x, const float* __restrict__ Km,
                    const float* __restrict__ Rm, float* __restrict__ out)
{
  __shared__ unsigned int xS[2][16 * ROWW]; // bf16 pairs, [m][k]
  __shared__ unsigned int hS[2][16 * ROWW]; // bf16 pairs, [m][u]

  const int tid   = threadIdx.x;
  const int lane  = tid & 31;
  const int wave  = tid >> 5;
  const int laneN = lane & 15;
  const int hi    = lane >> 4;
  const int n0    = wave * 16;
  const int b0    = blockIdx.x * 16;
  const int t0    = blockIdx.y * Ln;

  // ---- loop-invariant B-matrix fragments (K and R columns n0..n0+15) ----
  // B 32x16 bf16 layout: dword vv holds K={2vv,2vv+1} (vv<4) / {2vv+8,2vv+9} (vv>=4), +8 for lanes 16-31
  Frag bk[4], br[4];
  {
    const int n = n0 + laneN;
#pragma unroll
    for (int f = 0; f < 4; ++f) {
#pragma unroll
      for (int vv = 0; vv < 8; ++vv) {
        const int k = f * 32 + hi * 8 + (vv & 3) * 2 + (vv >> 2) * 16;
        bk[f].u[vv] = pack2f(Km[k * Un + n], Km[(k + 1) * Un + n]);
        br[f].u[vv] = pack2f(Rm[k * Un + n], Rm[(k + 1) * Un + n]);
      }
    }
  }

  // ---- cooperative x-tile staging: thread -> 8 contiguous f32 of one row ----
  const int xm = tid >> 4;
  const int xk = (tid & 15) * 8;
  const float* xrow = x + ((size_t)(b0 + xm) * Tn + t0) * Un + xk;

  { // preload local t=0
    v4f a = __builtin_nontemporal_load((const v4f*)xrow);
    v4f c = __builtin_nontemporal_load((const v4f*)(xrow + 4));
    v4u p; p.x = pack2f(a.x, a.y); p.y = pack2f(a.z, a.w);
    p.z = pack2f(c.x, c.y); p.w = pack2f(c.z, c.w);
    *(v4u*)&xS[0][xm * ROWW + (xk >> 1)] = p;
  }
  for (int j = tid; j < 16 * ROWW; j += 256) hS[0][j] = 0u; // h0 = 0
  __syncthreads();

  int cur = 0;
#pragma unroll 1
  for (int tt = 0; tt < Ln; ++tt) {
    const int t = t0 + tt;

    // prefetch next timestep's x tile into registers (independent of recurrence)
    v4f pa{}, pb{};
    const bool more = (tt + 1 < Ln);
    if (more) {
      const float* src = xrow + (size_t)(tt + 1) * Un;
      pa = __builtin_nontemporal_load((const v4f*)src);
      pb = __builtin_nontemporal_load((const v4f*)(src + 4));
    }

    // A 16x32 bf16 layout: V0-3: K={0..7}+8*hi ; V4-7: K={16..23}+8*hi (pairs per dword)
    Frag ax[4], ah[4];
    {
      const unsigned int* xw = xS[cur];
      const unsigned int* hw = hS[cur];
#pragma unroll
      for (int f = 0; f < 4; ++f) {
        const int base = laneN * ROWW + f * 16 + hi * 4;
        ax[f].q[0] = *(const v4u*)&xw[base];
        ax[f].q[1] = *(const v4u*)&xw[base + 8];
        ah[f].q[0] = *(const v4u*)&hw[base];
        ah[f].q[1] = *(const v4u*)&hw[base + 8];
      }
    }

    // two independent 4-deep WMMA chains (x@K and h@R) -> pipelines better
    v8f acc0 = {}, acc1 = {};
#pragma unroll
    for (int f = 0; f < 4; ++f) {
      acc0 = __builtin_amdgcn_wmma_f32_16x16x32_bf16(false, ax[f].b, false, bk[f].b,
                                                     (short)0, acc0, false, false);
      acc1 = __builtin_amdgcn_wmma_f32_16x16x32_bf16(false, ah[f].b, false, br[f].b,
                                                     (short)0, acc1, false, false);
    }

    // C/D layout: VGPR r -> M = r + 8*hi, N = laneN
    __bf16* hn = (__bf16*)hS[cur ^ 1];
    float*  og = out + ((size_t)(b0 + hi * 8) * Tn + t) * Un + n0 + laneN;
#pragma unroll
    for (int r = 0; r < 8; ++r) {
      const float v = acc0[r] + acc1[r];
      __builtin_nontemporal_store(v, og + (size_t)r * Tn * Un);
      hn[(r + hi * 8) * (2 * ROWW) + n0 + laneN] = (__bf16)v;
    }

    if (more) {
      v4u p; p.x = pack2f(pa.x, pa.y); p.y = pack2f(pa.z, pa.w);
      p.z = pack2f(pb.x, pb.y); p.w = pack2f(pb.z, pb.w);
      *(v4u*)&xS[cur ^ 1][xm * ROWW + (xk >> 1)] = p;
    }
    cur ^= 1;
    __syncthreads();
  }
}

// ---------------------------------------------------------------------------
// Pass B0: Mout = A @ A  (128x128 f32). Used to build M = R^32 (5 squarings).
// grid = 8 blocks x 256 threads; thread -> 8 contiguous outputs of one row.
// ---------------------------------------------------------------------------
__global__ __launch_bounds__(256)
void matsq128(const float* __restrict__ A, float* __restrict__ Mout)
{
  const int gtid = blockIdx.x * 256 + threadIdx.x;
  const int m  = gtid >> 4;
  const int ub = (gtid & 15) * 8;
  float acc[8] = {};
  for (int k = 0; k < Un; ++k) {
    const float a = A[m * Un + k];
    const float* brow = A + k * Un + ub;
#pragma unroll
    for (int i = 0; i < 8; ++i) acc[i] = fmaf(a, brow[i], acc[i]);
  }
#pragma unroll
  for (int i = 0; i < 8; ++i) Mout[m * Un + ub + i] = acc[i];
}

// ---------------------------------------------------------------------------
// Pass B1: sequential boundary scan across chunks (63 cheap steps).
//   H_c = S_{c-1} + H_{c-1} @ M,  S_{c-1} = uncorrected out[:, c*L-1, :].
// grid = 16 blocks (batch tiles); keeps H_{c-1} tile in LDS (f32).
// ---------------------------------------------------------------------------
__global__ __launch_bounds__(256, 1)
void boundary_scan(const float* __restrict__ out, const float* __restrict__ M,
                   float* __restrict__ H)
{
  __shared__ float Hprev[16 * Un];
  const int tid = threadIdx.x;
  const int b0  = blockIdx.x * 16;
  const int m   = tid >> 4;
  const int ub  = (tid & 15) * 8;

  for (int j = tid; j < 16 * Un; j += 256) Hprev[j] = 0.0f;
  __syncthreads();

  for (int c = 1; c < Cn; ++c) {
    float acc[8];
    const float* srow = out + ((size_t)(b0 + m) * Tn + (c * Ln - 1)) * Un + ub;
#pragma unroll
    for (int i = 0; i < 8; ++i) acc[i] = srow[i];
    for (int k = 0; k < Un; ++k) {
      const float hv = Hprev[m * Un + k];
      const float* mrow = M + k * Un + ub;
#pragma unroll
      for (int i = 0; i < 8; ++i) acc[i] = fmaf(hv, mrow[i], acc[i]);
    }
    __syncthreads();
#pragma unroll
    for (int i = 0; i < 8; ++i) {
      Hprev[m * Un + ub + i] = acc[i];
      H[(((size_t)c * Bn) + b0 + m) * Un + ub + i] = acc[i];
    }
    __syncthreads();
  }
}

// ---------------------------------------------------------------------------
// Pass C: fix-up. carry = H_c; per step carry = carry@R (WMMA); out[t] += carry.
// grid = 16 x 63 (chunk 0 needs no correction).
// ---------------------------------------------------------------------------
__global__ __launch_bounds__(256, 1)
void rnn_chunk_fix(const float* __restrict__ H, const float* __restrict__ Rm,
                   float* __restrict__ out)
{
  __shared__ unsigned int cS[2][16 * ROWW]; // carry, bf16 pairs

  const int tid   = threadIdx.x;
  const int lane  = tid & 31;
  const int wave  = tid >> 5;
  const int laneN = lane & 15;
  const int hi    = lane >> 4;
  const int n0    = wave * 16;
  const int b0    = blockIdx.x * 16;
  const int c     = blockIdx.y + 1;
  const int t0    = c * Ln;

  Frag br[4];
  {
    const int n = n0 + laneN;
#pragma unroll
    for (int f = 0; f < 4; ++f)
#pragma unroll
      for (int vv = 0; vv < 8; ++vv) {
        const int k = f * 32 + hi * 8 + (vv & 3) * 2 + (vv >> 2) * 16;
        br[f].u[vv] = pack2f(Rm[k * Un + n], Rm[(k + 1) * Un + n]);
      }
  }

  { // stage H_c tile as initial carry
    const int m  = tid >> 4;
    const int ub = (tid & 15) * 8;
    const float* src = H + (((size_t)c * Bn) + b0 + m) * Un + ub;
    v4f a = *(const v4f*)src;
    v4f d = *(const v4f*)(src + 4);
    v4u p; p.x = pack2f(a.x, a.y); p.y = pack2f(a.z, a.w);
    p.z = pack2f(d.x, d.y); p.w = pack2f(d.z, d.w);
    *(v4u*)&cS[0][m * ROWW + (ub >> 1)] = p;
  }
  __syncthreads();

  int cur = 0;
#pragma unroll 1
  for (int tt = 0; tt < Ln; ++tt) {
    const int t = t0 + tt;

    Frag ac[4];
    {
      const unsigned int* cw = cS[cur];
#pragma unroll
      for (int f = 0; f < 4; ++f) {
        const int base = laneN * ROWW + f * 16 + hi * 4;
        ac[f].q[0] = *(const v4u*)&cw[base];
        ac[f].q[1] = *(const v4u*)&cw[base + 8];
      }
    }

    v8f acc = {};
#pragma unroll
    for (int f = 0; f < 4; ++f)
      acc = __builtin_amdgcn_wmma_f32_16x16x32_bf16(false, ac[f].b, false, br[f].b,
                                                    (short)0, acc, false, false);

    __bf16* cn = (__bf16*)cS[cur ^ 1];
    float*  og = out + ((size_t)(b0 + hi * 8) * Tn + t) * Un + n0 + laneN;
#pragma unroll
    for (int r = 0; r < 8; ++r) {
      const float v = og[(size_t)r * Tn * Un] + acc[r];
      og[(size_t)r * Tn * Un] = v;
      cn[(r + hi * 8) * (2 * ROWW) + n0 + laneN] = (__bf16)acc[r];
    }
    cur ^= 1;
    __syncthreads();
  }
}

// ---------------------------------------------------------------------------
extern "C" void kernel_launch(void* const* d_in, const int* in_sizes, int n_in,
                              void* d_out, int out_size, void* d_ws, size_t ws_size,
                              hipStream_t stream) {
  (void)in_sizes; (void)n_in; (void)out_size; (void)ws_size;
  const float* x = (const float*)d_in[0];
  const float* K = (const float*)d_in[1];
  const float* R = (const float*)d_in[2];
  float* out = (float*)d_out;

  float* ws = (float*)d_ws;
  float* M0 = ws;                      // 128*128 f32
  float* M1 = ws + Un * Un;            // 128*128 f32
  float* H  = ws + 2 * Un * Un;        // Cn*Bn*Un f32 (~8 MB), slot 0 unused

  // Pass A: all chunk-local scans in parallel (heavy, WMMA)
  rnn_chunk_scan<<<dim3(Bn / 16, Cn), dim3(256), 0, stream>>>(x, K, R, out);

  // Pass B0: M = R^32 by repeated squaring (tiny kernels)
  matsq128<<<dim3(8), dim3(256), 0, stream>>>(R,  M0); // R^2
  matsq128<<<dim3(8), dim3(256), 0, stream>>>(M0, M1); // R^4
  matsq128<<<dim3(8), dim3(256), 0, stream>>>(M1, M0); // R^8
  matsq128<<<dim3(8), dim3(256), 0, stream>>>(M0, M1); // R^16
  matsq128<<<dim3(8), dim3(256), 0, stream>>>(M1, M0); // R^32

  // Pass B1: sequential scan of chunk-boundary states (63 cheap steps)
  boundary_scan<<<dim3(Bn / 16), dim3(256), 0, stream>>>(out, M0, H);

  // Pass C: per-chunk correction (WMMA), chunks 1..63
  rnn_chunk_fix<<<dim3(Bn / 16, Cn - 1), dim3(256), 0, stream>>>(H, R, out);
}
